// Encoder_80590766342336
// MI455X (gfx1250) — compile-verified, matrix operand order checked by reading, not run on hardware
//
#include <hip/hip_runtime.h>

typedef __attribute__((ext_vector_type(16))) __bf16 v16bf;
typedef __attribute__((ext_vector_type(8)))  float  v8f;
typedef unsigned short u16t;
typedef unsigned int   u32t;

static constexpr int Bn = 64, Tn = 128, En = 256, Hn = 1024, Gn = 4096;

union FragAB { uint4 q[2]; v16bf v; };

__device__ __forceinline__ u16t f2bf(float f) {
    u32t u = __float_as_uint(f);
    u32t r = u + 0x7FFFu + ((u >> 16) & 1u);   // round-to-nearest-even
    return (u16t)(r >> 16);
}
__device__ __forceinline__ float sigmoidf(float x) { return 1.0f / (1.0f + __expf(-x)); }

// CDNA5 async copy global->LDS (one b128 per active lane), tracked by ASYNCcnt.
__device__ __forceinline__ void async_b128_to_lds(u32t lds_off, const void* gaddr) {
    asm volatile("global_load_async_to_lds_b128 %0, %1, off"
                 :: "v"(lds_off), "v"(gaddr) : "memory");
}
__device__ __forceinline__ void wait_async0() {
    asm volatile("s_wait_asynccnt 0x0" ::: "memory");
}
// Low 32 bits of a generic pointer to LDS == wave-relative LDS byte address.
__device__ __forceinline__ u32t lds_addr(const void* p) {
    return (u32t)(unsigned long long)p;
}

// Grid-wide split barrier for the persistent recurrence kernel.
__device__ __forceinline__ void grid_sync(u32t* cnt, u32t target) {
    __syncthreads();
    if (threadIdx.x == 0) {
        __hip_atomic_fetch_add(cnt, 1u, __ATOMIC_ACQ_REL, __HIP_MEMORY_SCOPE_AGENT);
        while (__hip_atomic_load(cnt, __ATOMIC_ACQUIRE, __HIP_MEMORY_SCOPE_AGENT) < target)
            __builtin_amdgcn_s_sleep(2);
    }
    __syncthreads();
}

// ---------------------------------------------------------------------------
__global__ void k_reset(u32t* p) {
    if (threadIdx.x == 0 && blockIdx.x == 0) *p = 0;
}

// fp32 [K,N] row-major  ->  bf16 [N,K] (K-contiguous) for WMMA B-operands
__global__ void k_convert_transpose(const float* __restrict__ src,
                                    u16t* __restrict__ dst, int K, int N) {
    int idx = blockIdx.x * blockDim.x + threadIdx.x;
    if (idx >= K * N) return;
    int k = idx / N, n = idx % N;
    dst[n * K + k] = f2bf(src[idx]);
}

// embedding gather -> bf16, time-major rows m = t*B + b
__global__ void k_gather_embed(const int* __restrict__ x,
                               const float* __restrict__ emb,
                               u16t* __restrict__ eb) {
    int idx = blockIdx.x * blockDim.x + threadIdx.x;   // over T*B*E
    if (idx >= Tn * Bn * En) return;
    int e = idx & (En - 1);
    int m = idx >> 8;          // / E
    int t = m / Bn, b = m % Bn;
    int tok = x[b * Tn + t];
    eb[idx] = f2bf(emb[tok * En + e]);
}

// copy initial state (fp32) + optional bf16 mirror
__global__ void k_init_state(const float* __restrict__ src,
                             float* __restrict__ dstF,
                             u16t* __restrict__ dstB) {
    int idx = blockIdx.x * blockDim.x + threadIdx.x;
    if (idx >= Bn * Hn) return;
    float v = src[idx];
    dstF[idx] = v;
    if (dstB) dstB[idx] = f2bf(v);
}

// ---------------------------------------------------------------------------
// C[M,N] f32 = A_bf16[M,K] @ Bt_bf16[N,K]^T + bias[N]
// Block tile 64x128, 8 waves; async-to-LDS double-buffered A panel.
// sched_barrier keeps the 8 B-loads batched ahead of the 4 WMMAs.
// ---------------------------------------------------------------------------
__global__ __launch_bounds__(256) void k_gemm_bf16(
    const u16t* __restrict__ A, const u16t* __restrict__ Bt,
    const float* __restrict__ bias, float* __restrict__ C,
    int M, int N, int K) {
    __shared__ __align__(16) u16t shA[2][64 * 32];   // 2 x 4KB k-chunks

    const int nb   = N >> 7;
    const int rb   = blockIdx.x / nb;
    const int cb   = blockIdx.x % nb;
    const int tid  = threadIdx.x;
    const int w    = tid >> 5, lane = tid & 31;
    const int half = lane >> 4, mrow = lane & 15;
    const int rt   = w & 3;
    const int cg   = w >> 2;
    const int row0 = rb * 64 + rt * 16;
    const int colb = cb * 128 + cg * 64;

    v8f acc[4];
#pragma unroll
    for (int i = 0; i < 4; ++i) acc[i] = (v8f){0, 0, 0, 0, 0, 0, 0, 0};

    const int srow = tid >> 2;   // 0..63
    const int sseg = tid & 3;    // 0..3 (8 u16 each)
    const u16t* gA = A + (size_t)(rb * 64 + srow) * K + sseg * 8;
    const u32t  ldsA[2] = { lds_addr(&shA[0][srow * 32 + sseg * 8]),
                            lds_addr(&shA[1][srow * 32 + sseg * 8]) };

    const u16t* bB[4];
#pragma unroll
    for (int ct = 0; ct < 4; ++ct)
        bB[ct] = Bt + (size_t)(colb + ct * 16 + mrow) * K + half * 8;

    auto compute_chunk = [&](const u16t* buf, int kc) {
        FragAB b[4];
#pragma unroll
        for (int ct = 0; ct < 4; ++ct) {          // batch all 8 b128 loads
            b[ct].q[0] = *(const uint4*)(bB[ct] + kc);
            b[ct].q[1] = *(const uint4*)(bB[ct] + kc + 16);
        }
        FragAB a;
        const u16t* ap = buf + (rt * 16 + mrow) * 32 + half * 8;
        a.q[0] = *(const uint4*)(ap);
        a.q[1] = *(const uint4*)(ap + 16);
        __builtin_amdgcn_sched_barrier(0);        // loads stay ahead of WMMAs
#pragma unroll
        for (int ct = 0; ct < 4; ++ct)
            acc[ct] = __builtin_amdgcn_wmma_f32_16x16x32_bf16(
                false, a.v, false, b[ct].v, (short)0, acc[ct], false, false);
    };

    async_b128_to_lds(ldsA[0], gA);   // prologue: chunk 0 -> buffer 0
    wait_async0();
    __syncthreads();

    for (int kc = 0; kc < K; kc += 64) {
        if (kc + 32 < K) async_b128_to_lds(ldsA[1], gA + kc + 32);
        compute_chunk(shA[0], kc);
        if (kc + 32 < K) { wait_async0(); __syncthreads(); }
        if (kc + 64 < K) async_b128_to_lds(ldsA[0], gA + kc + 64);
        if (kc + 32 < K) compute_chunk(shA[1], kc + 32);
        if (kc + 64 < K) { wait_async0(); __syncthreads(); }
    }

#pragma unroll
    for (int ct = 0; ct < 4; ++ct) {
        int col = colb + ct * 16 + mrow;
        float bs = bias[col];
#pragma unroll
        for (int r = 0; r < 8; ++r) {
            int row = row0 + half * 8 + r;
            C[(size_t)row * N + col] = acc[ct][r] + bs;
        }
    }
}

// ---------------------------------------------------------------------------
// Persistent LSTM layer: 64 WGs run all T=128 steps, grid barrier per step.
// WG jt owns ALL 64 batch rows x 16 H-columns (=> 64 gate columns). Its U
// slice (64 x 1024 bf16 = 128KB) is async-staged into LDS ONCE and stays
// resident for the whole recurrence (CDNA5 320KB/WGP). Per step the full h
// matrix (64x1024 bf16 = 128KB, linear) is async-staged; K-loop runs entirely
// out of LDS with register-double-buffered fragments.
// 8 waves: gate g = w&3 (16 cols), row-tile pair rtp = w>>2 (rows 32*rtp..).
// LDS: shU 128KB + shH 128KB + shZ 16KB = 272KB.
// ---------------------------------------------------------------------------
__global__ __launch_bounds__(256) void k_lstm_layer(
    const float* __restrict__ xz,    // [T*B][G] precomputed x-projection+bias
    const u16t* __restrict__ Ut,     // [G][H] bf16, K-contiguous
    u16t* __restrict__ hbA,          // [B][H] bf16 ping (holds h0)
    u16t* __restrict__ hbB,          // [B][H] bf16 pong
    float* __restrict__ hstate,      // [B][H] f32 (d_out state region)
    float* __restrict__ cstate,      // [B][H] f32 running cell state
    u16t* __restrict__ y_bf,         // layer1: y1 base (time-major), else null
    float* __restrict__ y_f,         // layer2: d_out base [B,T,H], else null
    u32t* __restrict__ sync_cnt) {
    __shared__ __align__(16) u16t shU[64 * Hn];   // 128 KB, resident all steps
    __shared__ __align__(16) u16t shH[Bn * Hn];   // 128 KB, restaged per step
    __shared__ float shZ[4][Bn][16];              //  16 KB gate pre-activations

    const int jt   = blockIdx.x;          // 0..63 -> H cols [jt*16, jt*16+16)
    const int tid  = threadIdx.x;
    const int w    = tid >> 5, lane = tid & 31;
    const int half = lane >> 4, mrow = lane & 15;
    const int g    = w & 3;               // gate i,f,g,o
    const int rtp  = w >> 2;              // row-tile pair: rows 32*rtp..+31

    {   // stage this WG's U slice once: 8192 uint4, 32 per thread
        const u32t dst = lds_addr(shU);
#pragma unroll
        for (int i = 0; i < 32; ++i) {
            int fi = tid + i * 256;            // uint4 index
            int c  = fi >> 7;                  // local col 0..63
            int kq = fi & 127;                 // uint4 within K
            int n  = (c >> 4) * Hn + jt * 16 + (c & 15);   // global gate col
            async_b128_to_lds(dst + (u32t)fi * 16,
                              Ut + (size_t)n * Hn + (size_t)kq * 8);
        }
    }

    const u16t* ub = shU + ((g * 16 + mrow) << 10) + half * 8;
    const u16t* ab[2] = { shH + (((rtp * 2 + 0) * 16 + mrow) << 10) + half * 8,
                          shH + (((rtp * 2 + 1) * 16 + mrow) << 10) + half * 8 };

#pragma unroll 1
    for (int t = 0; t < Tn; ++t) {
        const u16t* hb_in  = (t & 1) ? hbB : hbA;
        u16t*       hb_out = (t & 1) ? hbA : hbB;

        {   // stage prev h (linear 128KB copy): 8192 uint4, 32 per thread
            const u32t dst = lds_addr(shH);
#pragma unroll
            for (int i = 0; i < 32; ++i) {
                int fi = tid + i * 256;
                async_b128_to_lds(dst + (u32t)fi * 16, hb_in + (size_t)fi * 8);
            }
            wait_async0();   // also covers the U staging on t==0
        }
        __syncthreads();

        v8f acc[2];
        acc[0] = (v8f){0, 0, 0, 0, 0, 0, 0, 0};
        acc[1] = (v8f){0, 0, 0, 0, 0, 0, 0, 0};

        FragAB aF[2][2], bF[2];
        auto load_frags = [&](int kc, int p) {
            bF[p].q[0] = *(const uint4*)(ub + kc);
            bF[p].q[1] = *(const uint4*)(ub + kc + 16);
#pragma unroll
            for (int j = 0; j < 2; ++j) {
                aF[p][j].q[0] = *(const uint4*)(ab[j] + kc);
                aF[p][j].q[1] = *(const uint4*)(ab[j] + kc + 16);
            }
        };

        load_frags(0, 0);
#pragma unroll
        for (int kc = 0; kc < Hn; kc += 32) {   // fully unrolled, regs rotate
            const int p = (kc >> 5) & 1, q = p ^ 1;
            if (kc + 32 < Hn) load_frags(kc + 32, q);
            acc[0] = __builtin_amdgcn_wmma_f32_16x16x32_bf16(
                false, aF[p][0].v, false, bF[p].v, (short)0, acc[0], false, false);
            acc[1] = __builtin_amdgcn_wmma_f32_16x16x32_bf16(
                false, aF[p][1].v, false, bF[p].v, (short)0, acc[1], false, false);
        }

        // add input projection, park gate slices in LDS
        const float* xz_t = xz + (size_t)t * Bn * Gn;
        const int colg = g * Hn + jt * 16 + mrow;
#pragma unroll
        for (int j = 0; j < 2; ++j) {
            int rt = rtp * 2 + j;
#pragma unroll
            for (int r = 0; r < 8; ++r) {
                int row = rt * 16 + half * 8 + r;
                shZ[g][row][mrow] = acc[j][r] + xz_t[(size_t)row * Gn + colg];
            }
        }
        __syncthreads();

        // elementwise cell update: 64 rows x 16 cols -> 4 per thread
#pragma unroll
        for (int p = 0; p < 4; ++p) {
            int idx  = tid + p * 256;
            int row  = idx >> 4;
            int jj   = idx & 15;
            int jcol = jt * 16 + jj;
            float zi = shZ[0][row][jj];
            float zf = shZ[1][row][jj];
            float zg = shZ[2][row][jj];
            float zo = shZ[3][row][jj];
            float c_old = cstate[row * Hn + jcol];
            float c_new = sigmoidf(zf) * c_old + sigmoidf(zi) * tanhf(zg);
            float h_new = sigmoidf(zo) * tanhf(c_new);
            cstate[row * Hn + jcol] = c_new;
            hstate[row * Hn + jcol] = h_new;
            hb_out[row * Hn + jcol] = f2bf(h_new);
            if (y_bf) y_bf[(size_t)(t * Bn + row) * Hn + jcol] = f2bf(h_new);
            if (y_f)  y_f[(size_t)row * Tn * Hn + (size_t)t * Hn + jcol] = h_new;
        }

        grid_sync(sync_cnt, (u32t)(t + 1) * 64);   // h visible to all WGs
    }
}

// ---------------------------------------------------------------------------
extern "C" void kernel_launch(void* const* d_in, const int* in_sizes, int n_in,
                              void* d_out, int out_size, void* d_ws, size_t ws_size,
                              hipStream_t stream) {
    const int*   x    = (const int*)d_in[0];
    const float* h0_1 = (const float*)d_in[1];
    const float* c0_1 = (const float*)d_in[2];
    const float* h0_2 = (const float*)d_in[3];
    const float* c0_2 = (const float*)d_in[4];
    const float* emb  = (const float*)d_in[5];
    const float* W1   = (const float*)d_in[6];
    const float* U1   = (const float*)d_in[7];
    const float* b1   = (const float*)d_in[8];
    const float* W2   = (const float*)d_in[9];
    const float* U2   = (const float*)d_in[10];
    const float* b2   = (const float*)d_in[11];

    float* out = (float*)d_out;
    float* h1o = out + (size_t)Bn * Tn * Hn;
    float* c1o = h1o + (size_t)Bn * Hn;
    float* h2o = c1o + (size_t)Bn * Hn;
    float* c2o = h2o + (size_t)Bn * Hn;

    char* wsp = (char*)d_ws;
    auto alloc = [&](size_t bytes) {
        char* p = wsp;
        wsp += (bytes + 255) & ~(size_t)255;
        return p;
    };
    u16t*  W1t  = (u16t*)alloc((size_t)Gn * En * 2);
    u16t*  U1t  = (u16t*)alloc((size_t)Gn * Hn * 2);
    u16t*  W2t  = (u16t*)alloc((size_t)Gn * Hn * 2);
    u16t*  U2t  = (u16t*)alloc((size_t)Gn * Hn * 2);
    u16t*  eb   = (u16t*)alloc((size_t)Tn * Bn * En * 2);
    u16t*  y1   = (u16t*)alloc((size_t)Tn * Bn * Hn * 2);
    u16t*  hb1a = (u16t*)alloc((size_t)Bn * Hn * 2);
    u16t*  hb1b = (u16t*)alloc((size_t)Bn * Hn * 2);
    u16t*  hb2a = (u16t*)alloc((size_t)Bn * Hn * 2);
    u16t*  hb2b = (u16t*)alloc((size_t)Bn * Hn * 2);
    u32t*  sync = (u32t*)alloc(256);
    float* xz   = (float*)alloc((size_t)Tn * Bn * Gn * 4);   // reused by both layers

    const int TPB = 256;
    auto blk = [&](long n) { return dim3((unsigned)((n + TPB - 1) / TPB)); };

    // 1) weights -> bf16, transposed K-contiguous
    k_convert_transpose<<<blk((long)En * Gn), TPB, 0, stream>>>(W1, W1t, En, Gn);
    k_convert_transpose<<<blk((long)Hn * Gn), TPB, 0, stream>>>(U1, U1t, Hn, Gn);
    k_convert_transpose<<<blk((long)Hn * Gn), TPB, 0, stream>>>(W2, W2t, Hn, Gn);
    k_convert_transpose<<<blk((long)Hn * Gn), TPB, 0, stream>>>(U2, U2t, Hn, Gn);

    // 2) embedding gather (time-major)
    k_gather_embed<<<blk((long)Tn * Bn * En), TPB, 0, stream>>>(x, emb, eb);

    // 3) initial states
    k_init_state<<<blk((long)Bn * Hn), TPB, 0, stream>>>(h0_1, h1o, hb1a);
    k_init_state<<<blk((long)Bn * Hn), TPB, 0, stream>>>(c0_1, c1o, nullptr);
    k_init_state<<<blk((long)Bn * Hn), TPB, 0, stream>>>(h0_2, h2o, hb2a);
    k_init_state<<<blk((long)Bn * Hn), TPB, 0, stream>>>(c0_2, c2o, nullptr);

    const int M = Tn * Bn;                       // 8192
    const dim3 gemm_grid((M / 64) * (Gn / 128)); // 4096 WGs

    // 4) layer-1 input projections: xz = eb @ W1 + b1
    k_gemm_bf16<<<gemm_grid, TPB, 0, stream>>>(eb, W1t, b1, xz, M, Gn, En);

    // 5) layer-1 recurrence (single persistent launch, 128 steps)
    k_reset<<<dim3(1), dim3(64), 0, stream>>>(sync);
    k_lstm_layer<<<dim3(64), TPB, 0, stream>>>(
        xz, U1t, hb1a, hb1b, h1o, c1o, y1, nullptr, sync);

    // 6) layer-2 input projections: xz = y1 @ W2 + b2 (xz buffer reused)
    k_gemm_bf16<<<gemm_grid, TPB, 0, stream>>>(y1, W2t, b2, xz, M, Gn, Hn);

    // 7) layer-2 recurrence, writing d_out [B,T,H] directly
    k_reset<<<dim3(1), dim3(64), 0, stream>>>(sync);
    k_lstm_layer<<<dim3(64), TPB, 0, stream>>>(
        xz, U2t, hb2a, hb2b, h2o, c2o, nullptr, out, sync);
}
